// PeriodicConvolution_36309653520733
// MI455X (gfx1250) — matrix operand order repaired and checked by reading, not compiled, not run on hardware
//
#include <hip/hip_runtime.h>
#include <math.h>

typedef _Float16 v16h __attribute__((ext_vector_type(16)));
typedef float    v8f  __attribute__((ext_vector_type(8)));

#define PC_NBASIS   16
#define PC_MAXR     3.5f
#define PC_KTOT     304          // (16 basis + 3 rhat) * 16 in_ch
#define PC_NCHUNK   10           // K padded to 320 = 10 * 32
#define WAVES_PER_BLOCK 8

// One wave processes one tile of 16 edges:
//   Z[16 x 320] (f16, built per-lane in registers, matching the CDNA5
//   16-bit A 16x32 VGPR layout) x Wmat[320 x 16] (f16 B fragments held
//   in registers) accumulated in f32 via 10 chained v_wmma_f32_16x16x32_f16.
__global__ __launch_bounds__(256) void
PeriodicConvolution_36309653520733_kernel(const float* __restrict__ x,
                                          const float* __restrict__ W,
                                          const float* __restrict__ evec,
                                          const int*   __restrict__ esrc,
                                          const int*   __restrict__ edst,
                                          float*       __restrict__ out,
                                          int E)
{
    const int lane = threadIdx.x & 31;
    const int wv   = threadIdx.x >> 5;
    const int N    = lane & 15;   // out-channel (B/C/D column)
    const int g    = lane >> 4;   // K-group select / C row-half / i-half
    const int nTiles = (E + 15) >> 4;

    // ---- Build B fragments once per wave: Wmat[k][o] = W[f][o][i], k = f*16+i,
    // padded with zeros for k in [304, 320). B 32x16 f16 layout: lane -> N = lane&15,
    // K = 8*(lane>>4) + t for t<8, 16 + 8*(lane>>4) + (t-8) for t>=8, elements packed 2 per VGPR (v16h element t).
    v16h b[PC_NCHUNK];
    #pragma unroll
    for (int kc = 0; kc < PC_NCHUNK; ++kc) {
        #pragma unroll
        for (int t = 0; t < 16; ++t) {
            const int k = kc * 32 + ((t < 8) ? (g * 8 + t) : (16 + g * 8 + (t - 8)));
            float w = 0.0f;
            if (k < PC_KTOT) {
                const int f = k >> 4;
                const int i = k & 15;
                w = W[f * 256 + N * 16 + i];   // W[f][o=N][i]
            }
            b[kc][t] = (_Float16)w;
        }
    }

    const float scale     = 0.20412414523193154f;      // 1/sqrt(24)
    const float inv_gamma = (float)PC_NBASIS / PC_MAXR; // 1/GAMMA
    const int   gw = blockIdx.x * WAVES_PER_BLOCK + wv;
    const int   nw = gridDim.x * WAVES_PER_BLOCK;

    for (int tile = gw; tile < nTiles; tile += nw) {
        const int base = tile << 4;
        const int e    = base + (lane & 15);   // this lane's edge
        // ---- per-edge inputs (lanes L and L+16 share an edge, split i-halves)
        float vx = 0.f, vy = 0.f, vz = 0.f;
        float xs[8];
        #pragma unroll
        for (int j = 0; j < 8; ++j) xs[j] = 0.f;
        if (e < E) {
            vx = evec[e * 3 + 0];
            vy = evec[e * 3 + 1];
            vz = evec[e * 3 + 2];
            const int s = esrc[e];
            const float4* xp = (const float4*)(x + (size_t)s * 16 + g * 8);
            const float4 x0 = xp[0];
            const float4 x1 = xp[1];
            xs[0] = x0.x; xs[1] = x0.y; xs[2] = x0.z; xs[3] = x0.w;
            xs[4] = x1.x; xs[5] = x1.y; xs[6] = x1.z; xs[7] = x1.w;
        }

        // ---- kernel features: 16 gaussian radial basis + unit direction
        const float r = sqrtf(vx * vx + vy * vy + vz * vz);
        float feat[20];
        #pragma unroll
        for (int f = 0; f < 16; ++f) {
            const float c = (float)f * (PC_MAXR / 15.0f);  // linspace(0, 3.5, 16)
            const float d = (r - c) * inv_gamma;
            feat[f] = __expf(-d * d);
        }
        const float ir = (r > 1e-10f) ? (1.0f / r) : 0.0f;
        feat[16] = vx * ir;
        feat[17] = vy * ir;
        feat[18] = vz * ir;
        feat[19] = 0.0f;                                   // K padding row

        // ---- 10 chained WMMAs over the K dimension
        v8f acc = {0.f, 0.f, 0.f, 0.f, 0.f, 0.f, 0.f, 0.f};
        #pragma unroll
        for (int kc = 0; kc < PC_NCHUNK; ++kc) {
            const float f0 = feat[2 * kc];
            const float f1 = feat[2 * kc + 1];
            // A fragment, 16-bit A 16x32 layout: this lane holds row (lane&15),
            // K = {8g..8g+7} in elements 0..7 and {16+8g..16+8g+7} in 8..15,
            // which is exactly (f_rel=0, i=8g+j) and (f_rel=1, i=8g+j).
            v16h a;
            #pragma unroll
            for (int j = 0; j < 8; ++j) {
                a[j]     = (_Float16)(f0 * xs[j]);
                a[8 + j] = (_Float16)(f1 * xs[j]);
            }
            acc = __builtin_amdgcn_wmma_f32_16x16x32_f16(
                /*neg_a=*/false, a, /*neg_b=*/false, b[kc],
                /*c_mod=*/(short)0, acc, /*reuse_a=*/false, /*reuse_b=*/false);
        }

        // ---- scatter-add: C/D layout: VGPR v, lane -> (M = v + 8g, N = lane&15)
        #pragma unroll
        for (int v = 0; v < 8; ++v) {
            const int m  = v + 8 * g;
            const int ee = base + m;
            if (ee < E) {
                const int d = edst[ee];
                atomicAdd(out + (size_t)d * 16 + N, acc[v] * scale);
            }
        }
    }
}

extern "C" void kernel_launch(void* const* d_in, const int* in_sizes, int n_in,
                              void* d_out, int out_size, void* d_ws, size_t ws_size,
                              hipStream_t stream) {
    const float* x    = (const float*)d_in[0];
    const float* W    = (const float*)d_in[1];
    const float* evec = (const float*)d_in[2];
    const int*   esrc = (const int*)d_in[3];
    const int*   edst = (const int*)d_in[4];
    float*       out  = (float*)d_out;

    const int E = in_sizes[3];               // number of edges

    // zero the accumulation buffer (capturable memset node)
    hipMemsetAsync(d_out, 0, (size_t)out_size * sizeof(float), stream);

    const int nTiles = (E + 15) >> 4;
    int blocks = (nTiles + WAVES_PER_BLOCK - 1) / WAVES_PER_BLOCK;
    if (blocks < 1) blocks = 1;

    PeriodicConvolution_36309653520733_kernel<<<blocks, 32 * WAVES_PER_BLOCK, 0, stream>>>(
        x, W, evec, esrc, edst, out, E);
}
